// Transformer_60894046322856
// MI455X (gfx1250) — compile-verified, matrix operand order checked by reading, not run on hardware
//
#include <hip/hip_runtime.h>
#include <hip/hip_bf16.h>

// MI455X / gfx1250 implementation.
// All dense GEMMs run on v_wmma_f32_16x16x32_f16 (wave32 WMMA), operands cast
// to f16 with f32 accumulation. GEMM A-tiles are staged into LDS via the
// Tensor Data Mover (tensor_load_to_lds + s_wait_tensorcnt) when the builtin
// is available, else via cooperative b128 copies. A never-taken conditional
// LDS store keeps the compiler from treating the TDM-written tile as undef
// while preserving addrspace(3) provenance (real ds_load_b128, not flat).
// Graph scatter + global kn^T v reduction use VALU + atomics (L2-resident).

typedef __attribute__((ext_vector_type(16))) _Float16 v16h;
typedef __attribute__((ext_vector_type(8)))  float    v8f;
typedef unsigned int __attribute__((ext_vector_type(4))) v4u;
typedef int  __attribute__((ext_vector_type(4))) v4i;
typedef int  __attribute__((ext_vector_type(8))) v8i;

#define HIDC 128
#define INC  256

#if defined(__gfx1250__) && __has_builtin(__builtin_amdgcn_tensor_load_to_lds) && \
    __has_builtin(__builtin_amdgcn_s_wait_tensorcnt)
#define HAVE_TDM 1
#else
#define HAVE_TDM 0
#endif

// ---------------------------------------------------------------------------
// helpers
// ---------------------------------------------------------------------------
__device__ __forceinline__ v16h load_frag16(const _Float16* p) {
  v16h f;
  ((uint4*)&f)[0] = *(const uint4*)(p);
  ((uint4*)&f)[1] = *(const uint4*)(p + 16);
  return f;
}

// ---------------------------------------------------------------------------
// WMMA GEMM: out[r][c] = sum_k A[r][k] * B[k][c] + bias[c]
// A: [nRows, K] f16 row-major (nRows % 32 == 0)
// BT: [128, K] f16 row-major with BT[n][k] = B[k][n]
// block: 256 threads = 8 waves; block tile = 32 rows x 128 cols
// wave w owns cols [16w, 16w+16), two 16x16 accumulators (rows +0, +16)
// A tile staged to LDS by the Tensor Data Mover (TENSORcnt) when available.
// ---------------------------------------------------------------------------
__global__ __launch_bounds__(256) void gemm_wmma_kernel(
    const _Float16* __restrict__ A, const _Float16* __restrict__ BT,
    const float* __restrict__ bias, float* __restrict__ out,
    int nRows, int K)
{
  __shared__ __attribute__((aligned(16))) _Float16 aLds[32 * INC];
  const int tid  = threadIdx.x;
  const int wave = tid >> 5;
  const int lane = tid & 31;
  const int r0   = blockIdx.x * 32;

#if HAVE_TDM
  if (wave == 0) {
    // Tensor DMA descriptor: 1-D tile of T = 32*K f16 elements,
    // global A + r0*K  ->  LDS offset 0 (aLds is the only LDS object).
    const unsigned T  = 32u * (unsigned)K;                    // elements
    const unsigned long long ga =
        (unsigned long long)(const void*)(A + (size_t)r0 * K);
    // group0: count=1 | lds_addr | global_addr[56:0] | type=2 (bits 127:126)
    v4u g0 = { 1u, 0u, (unsigned)ga,
               (unsigned)((ga >> 32) & 0x01ffffffu) | (2u << 30) };
    // group1: data_size=2B (code 1) @bits17:16; tensor_dim0=T @79:48;
    // tensor_dim1=1 @111:80; tile_dim0=T @127:112; tile_dim1/2=0;
    // tensor_dim0_stride=T @207:160; tensor_dim1_stride=0.
    v8i g1 = { (int)0x00010000u,
               (int)((T & 0xffffu) << 16),
               (int)((T >> 16) | (1u << 16)),
               (int)(T << 16),
               0,
               (int)T,
               0, 0 };
    v4i gz = { 0, 0, 0, 0 };
#if __clang_major__ >= 23
    v8i gz8 = { 0, 0, 0, 0, 0, 0, 0, 0 };
    __builtin_amdgcn_tensor_load_to_lds(g0, g1, gz, gz, gz8, 0);
#else
    __builtin_amdgcn_tensor_load_to_lds(g0, g1, gz, gz, 0);
#endif
    __builtin_amdgcn_s_wait_tensorcnt(0);
    // Never taken at runtime (nRows > 0): forces alias analysis to assume the
    // LDS tile was written (the TDM write is invisible to the compiler) while
    // keeping addrspace(3) provenance so the k-loop uses ds_load_b128.
    if (nRows < 0) aLds[tid] = (_Float16)0.0f;
  }
#else
  { // cooperative 128-bit loads of the contiguous 32xK A tile
    const uint4* Ag = (const uint4*)(A + (size_t)r0 * K);
    uint4* Al = (uint4*)aLds;
    const int n4 = (32 * K) >> 3;
    for (int i = tid; i < n4; i += 256) Al[i] = Ag[i];
  }
#endif
  __syncthreads();

  const int colLane = (wave << 4) + (lane & 15);
  const int kOff    = (lane < 16) ? 0 : 8;   // ISA 16-bit A/B fragment K split
  const int rA      = (lane & 15);

  v8f c0 = {}; v8f c1 = {};
  const _Float16* bRow  = BT + (size_t)colLane * K + kOff;
  const _Float16* aRow0 = aLds + rA * K + kOff;
  const _Float16* aRow1 = aLds + (rA + 16) * K + kOff;

  for (int kk = 0; kk < K; kk += 32) {
    v16h b  = load_frag16(bRow + kk);
    v16h a0 = load_frag16(aRow0 + kk);
    v16h a1 = load_frag16(aRow1 + kk);
    c0 = __builtin_amdgcn_wmma_f32_16x16x32_f16(false, a0, false, b, (short)0, c0, false, false);
    c1 = __builtin_amdgcn_wmma_f32_16x16x32_f16(false, a1, false, b, (short)0, c1, false, false);
  }

  const float bvv  = bias ? bias[colLane] : 0.0f;
  const int   rOff = (lane < 16) ? 0 : 8;    // f32 C layout: lanes>=16 -> M=e+8
  float* o0 = out + (size_t)(r0 + rOff) * HIDC + colLane;
#pragma unroll
  for (int e = 0; e < 8; ++e) {
    o0[(size_t)e * HIDC]        = c0[e] + bvv;
    o0[(size_t)(e + 16) * HIDC] = c1[e] + bvv;
  }
}

// ---------------------------------------------------------------------------
// small utility kernels
// ---------------------------------------------------------------------------
__global__ void fill_zero_kernel(float* __restrict__ p, size_t n) {
  size_t i = (size_t)blockIdx.x * blockDim.x + threadIdx.x;
  if (i < n) p[i] = 0.0f;
}

__global__ void cast_f16_kernel(const float* __restrict__ in, _Float16* __restrict__ out, size_t n) {
  size_t i = (size_t)blockIdx.x * blockDim.x + threadIdx.x;
  if (i < n) out[i] = (_Float16)in[i];
}

// dst[c*R + r] = (f16) src[r*C + c]   (src is [R,C] row-major)
__global__ void castT_kernel(const float* __restrict__ src, _Float16* __restrict__ dst, int R, int C) {
  int i = blockIdx.x * blockDim.x + threadIdx.x;
  if (i < R * C) {
    int c = i / R, r = i - c * R;
    dst[i] = (_Float16)src[(size_t)r * C + c];
  }
}

__global__ void deg_kernel(const int* __restrict__ col, float* __restrict__ deg, int E) {
  int e = blockIdx.x * blockDim.x + threadIdx.x;
  if (e < E) atomicAdd(&deg[col[e]], 1.0f);
}

__global__ void val_kernel(const int* __restrict__ row, const int* __restrict__ col,
                           const float* __restrict__ deg, float* __restrict__ val, int E) {
  int e = blockIdx.x * blockDim.x + threadIdx.x;
  if (e >= E) return;
  float dc = deg[col[e]], dr = deg[row[e]];
  val[e] = (dc > 0.0f && dr > 0.0f) ? rsqrtf(dc) * rsqrtf(dr) : 0.0f;
}

__global__ void bn_elu_kernel(const float* __restrict__ in,
                              const float* __restrict__ g, const float* __restrict__ b,
                              const float* __restrict__ m, const float* __restrict__ v,
                              float* __restrict__ outf, _Float16* __restrict__ outh, size_t n) {
  size_t i = (size_t)blockIdx.x * blockDim.x + threadIdx.x;
  if (i >= n) return;
  int c = (int)(i & (HIDC - 1));
  float y = (in[i] - m[c]) * rsqrtf(v[c] + 1e-5f) * g[c] + b[c];
  y = y > 0.0f ? y : expm1f(y);
  outf[i] = y;
  outh[i] = (_Float16)y;
}

// one wave (32 lanes) per row of 128
__global__ __launch_bounds__(256) void rownorm_kernel(const float* __restrict__ in,
                                                      float* __restrict__ outf,
                                                      _Float16* __restrict__ outh, int nRows) {
  int row  = blockIdx.x * 8 + (threadIdx.x >> 5);
  int lane = threadIdx.x & 31;
  if (row >= nRows) return;
  float4 xv = ((const float4*)(in + (size_t)row * HIDC))[lane];
  float ss = xv.x * xv.x + xv.y * xv.y + xv.z * xv.z + xv.w * xv.w;
#pragma unroll
  for (int o = 16; o > 0; o >>= 1) ss += __shfl_xor(ss, o, 32);
  float inv = rsqrtf(ss);
  float4 y; y.x = xv.x * inv; y.y = xv.y * inv; y.z = xv.z * inv; y.w = xv.w * inv;
  ((float4*)(outf + (size_t)row * HIDC))[lane] = y;
  if (outh) {
    _Float16* p = outh + (size_t)row * HIDC + lane * 4;
    p[0] = (_Float16)y.x; p[1] = (_Float16)y.y; p[2] = (_Float16)y.z; p[3] = (_Float16)y.w;
  }
}

// column sums of [nRows,128] -> out[128] (block = 128 threads, 256 rows/block)
__global__ __launch_bounds__(128) void colsum_kernel(const float* __restrict__ in,
                                                     float* __restrict__ out, int nRows) {
  int c  = threadIdx.x;
  int r0 = blockIdx.x * 256;
  int rE = r0 + 256; if (rE > nRows) rE = nRows;
  float s = 0.0f;
  for (int r = r0; r < rE; ++r) s += in[(size_t)r * HIDC + c];
  atomicAdd(&out[c], s);
}

// kvs[m][d] += sum_n kn[n][m] * v[n][d] over this block's row chunk (256 rows)
__global__ __launch_bounds__(256) void kv_reduce_kernel(const float* __restrict__ kn,
                                                        const float* __restrict__ v,
                                                        float* __restrict__ kvs, int nRows) {
  __shared__ float kBuf[8][HIDC];
  __shared__ float vBuf[8][HIDC];
  const int d  = threadIdx.x & 127;
  const int mh = threadIdx.x >> 7;  // 0 or 1 -> m in [mh*64, mh*64+64)
  float acc[64];
#pragma unroll
  for (int i = 0; i < 64; ++i) acc[i] = 0.0f;

  const int r0 = blockIdx.x * 256;
  int rEnd = r0 + 256; if (rEnd > nRows) rEnd = nRows;

  for (int rb = r0; rb < rEnd; rb += 8) {
    __syncthreads();
    for (int i = threadIdx.x; i < 8 * HIDC; i += 256) {
      int rr = rb + (i >> 7);
      int cc = i & 127;
      kBuf[i >> 7][cc] = (rr < nRows) ? kn[(size_t)rr * HIDC + cc] : 0.0f;
      vBuf[i >> 7][cc] = (rr < nRows) ? v[(size_t)rr * HIDC + cc]  : 0.0f;
    }
    __syncthreads();
    for (int j = 0; j < 8; ++j) {
      float vv = vBuf[j][d];
      const float* kr = &kBuf[j][mh * 64];
#pragma unroll
      for (int m = 0; m < 64; ++m) acc[m] = fmaf(kr[m], vv, acc[m]);
    }
  }
#pragma unroll
  for (int m = 0; m < 64; ++m) atomicAdd(&kvs[(size_t)(mh * 64 + m) * HIDC + d], acc[m]);
}

// attn = (num + vsum) / (dot(qn_row, ksum) + N); one wave per row
__global__ __launch_bounds__(256) void attn_div_kernel(const float* __restrict__ num,
                                                       const float* __restrict__ qn,
                                                       const float* __restrict__ ksum,
                                                       const float* __restrict__ vsum,
                                                       float* __restrict__ out, int nRows, float nVal) {
  int row  = blockIdx.x * 8 + (threadIdx.x >> 5);
  int lane = threadIdx.x & 31;
  if (row >= nRows) return;
  float4 q  = ((const float4*)(qn + (size_t)row * HIDC))[lane];
  float4 ks = ((const float4*)ksum)[lane];
  float dp = q.x * ks.x + q.y * ks.y + q.z * ks.z + q.w * ks.w;
#pragma unroll
  for (int o = 16; o > 0; o >>= 1) dp += __shfl_xor(dp, o, 32);
  float inv = 1.0f / (dp + nVal);
  float4 nm = ((const float4*)(num + (size_t)row * HIDC))[lane];
  float4 vs = ((const float4*)vsum)[lane];
  float4 y; y.x = (nm.x + vs.x) * inv; y.y = (nm.y + vs.y) * inv;
  y.z = (nm.z + vs.z) * inv; y.w = (nm.w + vs.w) * inv;
  ((float4*)(out + (size_t)row * HIDC))[lane] = y;
}

// agg[col[e]][:] += val[e] * g[row[e]][:]; one wave per edge, 4 dims per lane
__global__ __launch_bounds__(256) void scatter_kernel(const int* __restrict__ row,
                                                      const int* __restrict__ col,
                                                      const float* __restrict__ val,
                                                      const float* __restrict__ g,
                                                      float* __restrict__ agg, int E) {
  int e = blockIdx.x * 8 + (threadIdx.x >> 5);
  int j = threadIdx.x & 31;
  if (e >= E) return;
  float  w  = val[e];
  float4 gv = ((const float4*)(g + (size_t)row[e] * HIDC))[j];
  float* a  = agg + (size_t)col[e] * HIDC + j * 4;
  atomicAdd(a + 0, gv.x * w);
  atomicAdd(a + 1, gv.y * w);
  atomicAdd(a + 2, gv.z * w);
  atomicAdd(a + 3, gv.w * w);
}

// out = ELU(LN(attn + agg + r + h)); one wave per row; writes f32 + f16
// (no __restrict__ on h/outf: they alias by design)
__global__ __launch_bounds__(256) void add_ln_elu_kernel(const float* __restrict__ attn,
                                                         const float* __restrict__ agg,
                                                         const float* __restrict__ r,
                                                         const float* h,
                                                         const float* __restrict__ gamma,
                                                         const float* __restrict__ beta,
                                                         float* outf, _Float16* __restrict__ outh,
                                                         int nRows) {
  int row  = blockIdx.x * 8 + (threadIdx.x >> 5);
  int lane = threadIdx.x & 31;
  if (row >= nRows) return;
  size_t base = (size_t)row * HIDC;
  float4 a = ((const float4*)(attn + base))[lane];
  float4 b = ((const float4*)(agg + base))[lane];
  float4 c = ((const float4*)(r + base))[lane];
  float4 d = ((const float4*)(h + base))[lane];
  float s0 = a.x + b.x + c.x + d.x;
  float s1 = a.y + b.y + c.y + d.y;
  float s2 = a.z + b.z + c.z + d.z;
  float s3 = a.w + b.w + c.w + d.w;
  float sum = s0 + s1 + s2 + s3;
#pragma unroll
  for (int o = 16; o > 0; o >>= 1) sum += __shfl_xor(sum, o, 32);
  float mu = sum * (1.0f / HIDC);
  float d0 = s0 - mu, d1 = s1 - mu, d2 = s2 - mu, d3 = s3 - mu;
  float sq = d0 * d0 + d1 * d1 + d2 * d2 + d3 * d3;
#pragma unroll
  for (int o = 16; o > 0; o >>= 1) sq += __shfl_xor(sq, o, 32);
  float inv = rsqrtf(sq * (1.0f / HIDC) + 1e-5f);
  float4 gm = ((const float4*)gamma)[lane];
  float4 bt = ((const float4*)beta)[lane];
  float y0 = d0 * inv * gm.x + bt.x; y0 = y0 > 0.f ? y0 : expm1f(y0);
  float y1 = d1 * inv * gm.y + bt.y; y1 = y1 > 0.f ? y1 : expm1f(y1);
  float y2 = d2 * inv * gm.z + bt.z; y2 = y2 > 0.f ? y2 : expm1f(y2);
  float y3 = d3 * inv * gm.w + bt.w; y3 = y3 > 0.f ? y3 : expm1f(y3);
  float4 y; y.x = y0; y.y = y1; y.z = y2; y.w = y3;
  ((float4*)(outf + base))[lane] = y;
  _Float16* p = outh + base + lane * 4;
  p[0] = (_Float16)y0; p[1] = (_Float16)y1; p[2] = (_Float16)y2; p[3] = (_Float16)y3;
}

// ---------------------------------------------------------------------------
// orchestration
// ---------------------------------------------------------------------------
extern "C" void kernel_launch(void* const* d_in, const int* in_sizes, int n_in,
                              void* d_out, int out_size, void* d_ws, size_t ws_size,
                              hipStream_t stream) {
  const float* x        = (const float*)d_in[0];
  const int*   edge     = (const int*)d_in[1];
  const float* mlp_W0   = (const float*)d_in[2];
  const float* mlp_b0   = (const float*)d_in[3];
  const float* mlp_W    = (const float*)d_in[4];
  const float* mlp_b    = (const float*)d_in[5];
  const float* bn_gamma = (const float*)d_in[6];
  const float* bn_beta  = (const float*)d_in[7];
  const float* bn_mean  = (const float*)d_in[8];
  const float* bn_var   = (const float*)d_in[9];
  const float* Wq = (const float*)d_in[10];  const float* bq = (const float*)d_in[11];
  const float* Wk = (const float*)d_in[12];  const float* bk = (const float*)d_in[13];
  const float* Wv = (const float*)d_in[14];  const float* bv = (const float*)d_in[15];
  const float* Wg = (const float*)d_in[16];  const float* bg = (const float*)d_in[17];
  const float* Wr = (const float*)d_in[18];  const float* br = (const float*)d_in[19];
  const float* ln_gamma = (const float*)d_in[20];
  const float* ln_beta  = (const float*)d_in[21];

  const int N = in_sizes[0] / INC;   // 100000 (multiple of 32)
  const int E = in_sizes[1] / 2;     // 3200000
  const int L = 3;

  // workspace carve-out (all 256B aligned)
  char* ws = (char*)d_ws;
  size_t off = 0;
  auto alloc = [&](size_t bytes) -> char* {
    char* p = ws + off;
    off += (bytes + 255) & ~(size_t)255;
    return p;
  };
  _Float16* xh   = (_Float16*)alloc((size_t)N * INC * 2);   // x f16; reused as qn_f32 later
  _Float16* hh   = (_Float16*)alloc((size_t)N * HIDC * 2);  // h f16 (GEMM A operand)
  float*    tmp  = (float*)   alloc((size_t)N * HIDC * 4);  // GEMM raw output
  _Float16* qh   = (_Float16*)alloc((size_t)N * HIDC * 2);  // qn f16
  float*    knb  = (float*)   alloc((size_t)N * HIDC * 4);  // kn f32; reused as g
  float*    vb   = (float*)   alloc((size_t)N * HIDC * 4);  // v f32; reused as attn_out
  float*    aggb = (float*)   alloc((size_t)N * HIDC * 4);  // GCN aggregate
  float*    deg  = (float*)   alloc((size_t)N * 4);
  float*    valb = (float*)   alloc((size_t)E * 4);
  float*    kvs  = (float*)   alloc((size_t)HIDC * HIDC * 4);
  _Float16* kvsT = (_Float16*)alloc((size_t)HIDC * HIDC * 2);
  float*    ksum = (float*)   alloc(HIDC * 4);
  float*    vsum = (float*)   alloc(HIDC * 4);
  _Float16* wT   = (_Float16*)alloc((size_t)INC * HIDC * 2); // transposed f16 weights (reused)

  float* hf  = (float*)d_out;       // f32 h lives in d_out (overwritten every call)
  float* qnf = (float*)xh;          // reuse x_f16 region after the MLP (same byte size)
  const int* erow = edge;
  const int* ecol = edge + E;

  const int eBlocks   = (E + 255) / 256;
  const int rowWBlks  = (N + 7) / 8;          // wave-per-row kernels
  const int gemmBlks  = N / 32;
  const int nhBlocks  = (int)(((size_t)N * HIDC + 255) / 256);

  // --- GCN edge normalization (shared across layers) ---
  fill_zero_kernel<<<(N + 255) / 256, 256, 0, stream>>>(deg, (size_t)N);
  deg_kernel<<<eBlocks, 256, 0, stream>>>(ecol, deg, E);
  val_kernel<<<eBlocks, 256, 0, stream>>>(erow, ecol, deg, valb, E);

  // --- input MLP ---
  cast_f16_kernel<<<(int)(((size_t)N * INC + 255) / 256), 256, 0, stream>>>(x, xh, (size_t)N * INC);
  castT_kernel<<<(INC * HIDC + 255) / 256, 256, 0, stream>>>(mlp_W0, wT, INC, HIDC);
  gemm_wmma_kernel<<<gemmBlks, 256, 0, stream>>>(xh, wT, mlp_b0, tmp, N, INC);
  bn_elu_kernel<<<nhBlocks, 256, 0, stream>>>(tmp, bn_gamma, bn_beta, bn_mean, bn_var,
                                              hf, hh, (size_t)N * HIDC);
  for (int i = 1; i < L; ++i) {
    castT_kernel<<<(HIDC * HIDC + 255) / 256, 256, 0, stream>>>(mlp_W + (size_t)(i - 1) * HIDC * HIDC,
                                                                wT, HIDC, HIDC);
    gemm_wmma_kernel<<<gemmBlks, 256, 0, stream>>>(hh, wT, mlp_b + (size_t)(i - 1) * HIDC, tmp, N, HIDC);
    bn_elu_kernel<<<nhBlocks, 256, 0, stream>>>(tmp, bn_gamma + i * HIDC, bn_beta + i * HIDC,
                                                bn_mean + i * HIDC, bn_var + i * HIDC,
                                                hf, hh, (size_t)N * HIDC);
  }

  // --- transformer layers ---
  for (int l = 0; l < L; ++l) {
    const size_t wOff = (size_t)l * HIDC * HIDC;
    const size_t bOff = (size_t)l * HIDC;

    // q -> qn (f32 + f16)
    castT_kernel<<<(HIDC * HIDC + 255) / 256, 256, 0, stream>>>(Wq + wOff, wT, HIDC, HIDC);
    gemm_wmma_kernel<<<gemmBlks, 256, 0, stream>>>(hh, wT, bq + bOff, tmp, N, HIDC);
    rownorm_kernel<<<rowWBlks, 256, 0, stream>>>(tmp, qnf, qh, N);

    // k -> kn (f32)
    castT_kernel<<<(HIDC * HIDC + 255) / 256, 256, 0, stream>>>(Wk + wOff, wT, HIDC, HIDC);
    gemm_wmma_kernel<<<gemmBlks, 256, 0, stream>>>(hh, wT, bk + bOff, tmp, N, HIDC);
    rownorm_kernel<<<rowWBlks, 256, 0, stream>>>(tmp, knb, (_Float16*)nullptr, N);

    // v
    castT_kernel<<<(HIDC * HIDC + 255) / 256, 256, 0, stream>>>(Wv + wOff, wT, HIDC, HIDC);
    gemm_wmma_kernel<<<gemmBlks, 256, 0, stream>>>(hh, wT, bv + bOff, vb, N, HIDC);

    // global reductions: ksum, vsum, kvs = kn^T v
    fill_zero_kernel<<<(HIDC * HIDC + 255) / 256, 256, 0, stream>>>(kvs, (size_t)HIDC * HIDC);
    fill_zero_kernel<<<1, 256, 0, stream>>>(ksum, (size_t)HIDC);
    fill_zero_kernel<<<1, 256, 0, stream>>>(vsum, (size_t)HIDC);
    colsum_kernel<<<(N + 255) / 256, 128, 0, stream>>>(knb, ksum, N);
    colsum_kernel<<<(N + 255) / 256, 128, 0, stream>>>(vb, vsum, N);
    kv_reduce_kernel<<<(N + 255) / 256, 256, 0, stream>>>(knb, vb, kvs, N);
    castT_kernel<<<(HIDC * HIDC + 255) / 256, 256, 0, stream>>>(kvs, kvsT, HIDC, HIDC);

    // num = qn @ kvs ; attn = (num + vsum) / (qn . ksum + N)   (attn into vb)
    gemm_wmma_kernel<<<gemmBlks, 256, 0, stream>>>(qh, kvsT, (const float*)nullptr, tmp, N, HIDC);
    attn_div_kernel<<<rowWBlks, 256, 0, stream>>>(tmp, qnf, ksum, vsum, vb, N, (float)N);

    // g (into knb), then edge scatter into aggb
    castT_kernel<<<(HIDC * HIDC + 255) / 256, 256, 0, stream>>>(Wg + wOff, wT, HIDC, HIDC);
    gemm_wmma_kernel<<<gemmBlks, 256, 0, stream>>>(hh, wT, bg + bOff, knb, N, HIDC);
    fill_zero_kernel<<<nhBlocks, 256, 0, stream>>>(aggb, (size_t)N * HIDC);
    scatter_kernel<<<(E + 7) / 8, 256, 0, stream>>>(erow, ecol, valb, knb, aggb, E);

    // r, then out = ELU(LN(attn + agg + r + h)) -> h (f32 in d_out, f16 copy)
    castT_kernel<<<(HIDC * HIDC + 255) / 256, 256, 0, stream>>>(Wr + wOff, wT, HIDC, HIDC);
    gemm_wmma_kernel<<<gemmBlks, 256, 0, stream>>>(hh, wT, br + bOff, tmp, N, HIDC);
    add_ln_elu_kernel<<<rowWBlks, 256, 0, stream>>>(vb, aggb, tmp, hf,
                                                    ln_gamma + bOff, ln_beta + bOff,
                                                    hf, hh, N);
  }
  // final h already lives in d_out (f32)
}
